// ConvZero_12017318494892
// MI455X (gfx1250) — matrix operand order repaired and checked by reading, not compiled, not run on hardware
//
#include <hip/hip_runtime.h>

#define N_NODES 100000
#define N_EDGES 640000
#define HDIM    128
#define H2DIM   256
#define EDIM    16
#define BPAD    40   // bf16 slots per (kchunk, column) group: 32 data + 8 pad -> bank-conflict-free

typedef __attribute__((ext_vector_type(16))) __bf16 v16bf;
typedef __attribute__((ext_vector_type(8)))  float  v8f;

static __device__ __forceinline__ v8f wmma_bf16(v16bf a, v16bf b, v8f c) {
  // 8 args: (neg_a, A, neg_b, B, c_mod, C, reuse_a, reuse_b)
  return __builtin_amdgcn_wmma_f32_16x16x32_bf16(false, a, false, b, (short)0, c, false, false);
}

// A fragment (16x32 bf16): lane l holds row M=l&15; element t <-> K = 8h + t + (t&8)
template<bool ACT>
static __device__ __forceinline__ v16bf load_a_frag(const float* __restrict__ A, int row0, int K, int k0,
                                                    const float* __restrict__ scale,
                                                    const float* __restrict__ shift) {
  const int lane = threadIdx.x & 31;
  const int m = lane & 15, h = lane >> 4;
  const float* rp = A + (size_t)(row0 + m) * K;
  v16bf a;
#pragma unroll
  for (int t = 0; t < 16; ++t) {
    int k = k0 + 8 * h + t + (t & 8);
    float x = rp[k];
    if (ACT) x = fmaxf(fmaf(x, scale[k], shift[k]), 0.0f);  // fused BN + ReLU
    a[t] = (__bf16)x;
  }
  return a;
}

// B fragment from swizzled LDS: layout [(kc*NOUT + n)*BPAD + (k&31)]; each lane's
// 16 elements (K = 8h + t + (t&8), fixed column n) are two 8-runs -> 2x ds_load_b128.
static __device__ __forceinline__ v16bf load_b_frag_sw(const __bf16* __restrict__ Wl, int nout, int kc, int n0) {
  typedef __attribute__((ext_vector_type(8))) __bf16 v8bf;
  const int lane = threadIdx.x & 31;
  const int n = n0 + (lane & 15), h = lane >> 4;
  const __bf16* p = Wl + (size_t)(kc * nout + n) * BPAD + h * 8;
  v8bf lo = *(const v8bf*)p;
  v8bf hi = *(const v8bf*)(p + 16);
  return __builtin_shufflevector(lo, hi, 0, 1, 2, 3, 4, 5, 6, 7, 8, 9, 10, 11, 12, 13, 14, 15);
}

// Stage W[K][NOUT] (f32, global) into swizzled bf16 LDS layout.
static __device__ __forceinline__ void stage_w_sw(__bf16* __restrict__ Wl, const float* __restrict__ W,
                                                  int K, int nout) {
  for (int i = threadIdx.x; i < K * nout; i += blockDim.x) {
    int k = i / nout, n = i - k * nout;
    Wl[(size_t)((k >> 5) * nout + n) * BPAD + (k & 31)] = (__bf16)W[i];
  }
}

// Generic GEMM: Out[M,NOUT] = act_in(A)[M,K] @ W[K,NOUT] (+bias); optional column stats of raw Out.
// One wave owns one 16-row M-tile; A fragments live in registers across all N-tiles.
// All epilogue accesses are base-pointer + compile-time-immediate offsets.
template<int K, int NOUT, bool ACT_IN, bool STATS, bool BIAS>
__global__ void gemm_kernel(const float* __restrict__ A, const float* __restrict__ W,
                            float* __restrict__ Out, int mrows,
                            const float* __restrict__ scale_in, const float* __restrict__ shift_in,
                            const float* __restrict__ bias,
                            float* __restrict__ sum_out, float* __restrict__ sq_out) {
  extern __shared__ char smem[];
  __bf16* Wl = (__bf16*)smem;
  stage_w_sw(Wl, W, K, NOUT);
  __syncthreads();

  const int lane = threadIdx.x & 31;
  const int m = lane & 15, h = lane >> 4;
  const int gwave  = blockIdx.x * (blockDim.x >> 5) + (threadIdx.x >> 5);
  const int wtotal = gridDim.x * (blockDim.x >> 5);
  const int mtiles = mrows / 16;

  const float* bp  = BIAS  ? (bias + m)    : nullptr;
  float*       sp  = STATS ? (sum_out + m) : nullptr;
  float*       qp  = STATS ? (sq_out + m)  : nullptr;

  for (int mt = gwave; mt < mtiles; mt += wtotal) {
    v16bf afr[K / 32];
#pragma unroll
    for (int kc = 0; kc < K / 32; ++kc)
      afr[kc] = load_a_frag<ACT_IN>(A, mt * 16, K, kc * 32, scale_in, shift_in);

    float* outp = Out + (size_t)(mt * 16 + 8 * h) * NOUT + m;   // one base per M-tile

#pragma unroll
    for (int nt = 0; nt < NOUT / 16; ++nt) {
      v8f c = {};
#pragma unroll
      for (int kc = 0; kc < K / 32; ++kc)
        c = wmma_bf16(afr[kc], load_b_frag_sw(Wl, NOUT, kc, nt * 16), c);

      if (BIAS) {
        float bb = bp[nt * 16];
#pragma unroll
        for (int r = 0; r < 8; ++r) c[r] += bb;
      }
      float s = 0.0f, q = 0.0f;
#pragma unroll
      for (int r = 0; r < 8; ++r) {
        float v = c[r];
        outp[r * NOUT + nt * 16] = v;                           // base + immediate
        if (STATS) { s += v; q += v * v; }
      }
      if (STATS) {
        s += __shfl_xor(s, 16);   // wave32: combine the two row-halves of each column
        q += __shfl_xor(q, 16);
        if (lane < 16) { atomicAdd(&sp[nt * 16], s); atomicAdd(&qp[nt * 16], q); }
      }
    }
  }
}

// Edge pass: msg = edge_rep@W3 + pad32(edge_attr)@We + be + P1[src] + P2[dst].
// One wave owns one 16-edge tile: edge_rep read exactly once (dominant HBM stream).
// STATS pass: column sum/sumsq over all E.  SCATTER pass: relu(bn(msg)) atomically added to y[dst].
template<bool SCATTER>
__global__ void edge_kernel(const float* __restrict__ edge_rep, const float* __restrict__ edge_attr,
                            const int* __restrict__ src, const int* __restrict__ dst,
                            const float* __restrict__ P1, const float* __restrict__ P2,
                            const float* __restrict__ W3, const float* __restrict__ We,
                            const float* __restrict__ be,
                            const float* __restrict__ scale1, const float* __restrict__ shift1,
                            float* __restrict__ sum_out, float* __restrict__ sq_out,
                            float* __restrict__ y) {
  extern __shared__ char smem[];
  __bf16* W3l = (__bf16*)smem;                         // 4 chunks * 128 cols * BPAD
  __bf16* Wel = W3l + 4 * HDIM * BPAD;                 // 1 chunk * 128 cols * BPAD (K 16..31 zero)
  stage_w_sw(W3l, W3, HDIM, HDIM);
  for (int i = threadIdx.x; i < 32 * HDIM; i += blockDim.x) {
    int k = i / HDIM, n = i - k * HDIM;
    Wel[(size_t)n * BPAD + k] = (k < EDIM) ? (__bf16)We[k * HDIM + n] : (__bf16)0.0f;
  }
  __syncthreads();

  const int lane = threadIdx.x & 31;
  const int m = lane & 15, h = lane >> 4;
  const int gwave  = blockIdx.x * (blockDim.x >> 5) + (threadIdx.x >> 5);
  const int wtotal = gridDim.x * (blockDim.x >> 5);
  const int ETILES = N_EDGES / 16;

  const float* bep = be + m;
  const float* scp = SCATTER ? (scale1 + m) : nullptr;
  const float* shp = SCATTER ? (shift1 + m) : nullptr;
  float*       sp  = SCATTER ? nullptr : (sum_out + m);
  float*       qp  = SCATTER ? nullptr : (sq_out + m);

  for (int et = gwave; et < ETILES; et += wtotal) {
    const int e0 = et * 16;
    v16bf afr[HDIM / 32];
#pragma unroll
    for (int kc = 0; kc < HDIM / 32; ++kc)
      afr[kc] = load_a_frag<false>(edge_rep, e0, HDIM, kc * 32, nullptr, nullptr);
    v16bf attr;   // edge encoder A: K=16 zero-padded to one K=32 chunk
    {
      const float* rp = edge_attr + (size_t)(e0 + m) * EDIM;
#pragma unroll
      for (int t = 0; t < 16; ++t) {
        int k = 8 * h + t + (t & 8);
        attr[t] = (k < EDIM) ? (__bf16)rp[k] : (__bf16)0.0f;
      }
    }
    // per-row base pointers: every later access is base + immediate offset
    const float* p1p[8];
    const float* p2p[8];
    float*       yp[8];
#pragma unroll
    for (int r = 0; r < 8; ++r) {
      int si = src[e0 + 8 * h + r];
      int di = dst[e0 + 8 * h + r];
      p1p[r] = P1 + (size_t)si * HDIM + m;
      p2p[r] = P2 + (size_t)di * HDIM + m;
      if (SCATTER) yp[r] = y + (size_t)di * HDIM + m;
    }

#pragma unroll
    for (int nt = 0; nt < HDIM / 16; ++nt) {
      v8f c = {};
#pragma unroll
      for (int kc = 0; kc < HDIM / 32; ++kc)
        c = wmma_bf16(afr[kc], load_b_frag_sw(W3l, HDIM, kc, nt * 16), c);
      c = wmma_bf16(attr, load_b_frag_sw(Wel, HDIM, 0, nt * 16), c);

      const float bb = bep[nt * 16];
#pragma unroll
      for (int r = 0; r < 8; ++r)
        c[r] += bb + p1p[r][nt * 16] + p2p[r][nt * 16];         // L2-resident gathers, imm offsets

      if (SCATTER) {
        const float sc = scp[nt * 16], sh = shp[nt * 16];
#pragma unroll
        for (int r = 0; r < 8; ++r) {
          float v = fmaxf(fmaf(c[r], sc, sh), 0.0f);
          atomicAdd(&yp[r][nt * 16], v);                        // y (51MB) is L2-resident
        }
      } else {
        float s = 0.0f, q = 0.0f;
#pragma unroll
        for (int r = 0; r < 8; ++r) { s += c[r]; q += c[r] * c[r]; }
        s += __shfl_xor(s, 16);
        q += __shfl_xor(q, 16);
        if (lane < 16) { atomicAdd(&sp[nt * 16], s); atomicAdd(&qp[nt * 16], q); }
      }
    }
  }
}

__global__ void bn_finalize_kernel(const float* __restrict__ sum, const float* __restrict__ sq,
                                   const float* __restrict__ g, const float* __restrict__ b,
                                   float inv_cnt, int C,
                                   float* __restrict__ scale, float* __restrict__ shift) {
  int i = blockIdx.x * blockDim.x + threadIdx.x;
  if (i < C) {
    float mu  = sum[i] * inv_cnt;
    float var = sq[i] * inv_cnt - mu * mu;      // biased variance, matches jnp.var
    float s = g[i] * rsqrtf(var + 1e-5f);
    scale[i] = s;
    shift[i] = fmaf(-mu, s, b[i]);
  }
}

__global__ void zero_kernel(float* __restrict__ p, size_t n) {
  size_t i = (size_t)blockIdx.x * blockDim.x + threadIdx.x;
  size_t stride = (size_t)gridDim.x * blockDim.x;
  for (; i < n; i += stride) p[i] = 0.0f;
}

static inline int imin(int a, int b) { return a < b ? a : b; }

extern "C" void kernel_launch(void* const* d_in, const int* in_sizes, int n_in,
                              void* d_out, int out_size, void* d_ws, size_t ws_size,
                              hipStream_t stream) {
  const float* node_rep  = (const float*)d_in[0];
  const float* edge_rep  = (const float*)d_in[1];
  const float* edge_attr = (const float*)d_in[2];
  const int*   eidx      = (const int*)  d_in[3];
  const float* W1  = (const float*)d_in[4];
  const float* W2  = (const float*)d_in[5];
  const float* W3  = (const float*)d_in[6];
  const float* We  = (const float*)d_in[7];
  const float* be  = (const float*)d_in[8];
  const float* bn_g = (const float*)d_in[9];
  const float* bn_b = (const float*)d_in[10];
  const float* Wm1 = (const float*)d_in[11];
  const float* g1  = (const float*)d_in[12];
  const float* b1  = (const float*)d_in[13];
  const float* Wm2 = (const float*)d_in[14];
  const float* g2  = (const float*)d_in[15];
  const float* b2  = (const float*)d_in[16];
  const float* Wm3 = (const float*)d_in[17];
  const float* bm3 = (const float*)d_in[18];

  const int* src = eidx;
  const int* dst = eidx + N_EDGES;

  float* ws = (float*)d_ws;
  const size_t NH  = (size_t)N_NODES * HDIM;
  const size_t N2H = (size_t)N_NODES * H2DIM;
  float* P1 = ws;              // node @ W1
  float* P2 = P1 + NH;         // node @ W2
  float* Y  = P2 + NH;         // scattered messages
  float* H1 = Y  + NH;         // y @ Wm1
  float* H2 = H1 + N2H;        // act(H1) @ Wm2
  float* S  = H2 + N2H;        // small stat/scale arrays
  float* sum1 = S;              float* sq1 = sum1 + HDIM;
  float* sum2 = sq1 + HDIM;     float* sq2 = sum2 + H2DIM;
  float* sum3 = sq2 + H2DIM;    float* sq3 = sum3 + H2DIM;
  float* scale1 = sq3 + H2DIM;  float* shift1 = scale1 + HDIM;
  float* scale2 = shift1 + HDIM;  float* shift2 = scale2 + H2DIM;
  float* scale3 = shift2 + H2DIM; float* shift3 = scale3 + H2DIM;

  const int THREADS = 256;  // 8 wave32 per block

  // dynamic-LDS sizes for swizzled bf16 weight tiles
  const size_t lds_128_128 = (size_t)(HDIM  / 32) * HDIM  * BPAD * sizeof(__bf16);  //  40 KB
  const size_t lds_128_256 = (size_t)(HDIM  / 32) * H2DIM * BPAD * sizeof(__bf16);  //  80 KB
  const size_t lds_256_256 = (size_t)(H2DIM / 32) * H2DIM * BPAD * sizeof(__bf16);  // 160 KB
  const size_t lds_256_128 = (size_t)(H2DIM / 32) * HDIM  * BPAD * sizeof(__bf16);  //  80 KB
  const size_t lds_edge    = (size_t)(4 * HDIM + HDIM) * BPAD * sizeof(__bf16);     //  50 KB

  (void)hipFuncSetAttribute((const void*)gemm_kernel<HDIM,  H2DIM, false, true,  false>,
                            hipFuncAttributeMaxDynamicSharedMemorySize, (int)lds_128_256);
  (void)hipFuncSetAttribute((const void*)gemm_kernel<H2DIM, H2DIM, true,  true,  false>,
                            hipFuncAttributeMaxDynamicSharedMemorySize, (int)lds_256_256);
  (void)hipFuncSetAttribute((const void*)gemm_kernel<H2DIM, HDIM,  true,  false, true>,
                            hipFuncAttributeMaxDynamicSharedMemorySize, (int)lds_256_128);

  zero_kernel<<<1024, THREADS, 0, stream>>>(Y, NH);
  zero_kernel<<<8, THREADS, 0, stream>>>(sum1, (size_t)(2 * HDIM + 4 * H2DIM));

  const int mtiles   = N_NODES / 16;                       // 6250
  const int gblocks  = imin((mtiles + 7) / 8, 2048);
  const int eblocks  = imin((N_EDGES / 16 + 7) / 8, 2048);

  // node projections P1, P2
  gemm_kernel<HDIM, HDIM, false, false, false><<<gblocks, THREADS, lds_128_128, stream>>>(
      node_rep, W1, P1, N_NODES, nullptr, nullptr, nullptr, nullptr, nullptr);
  gemm_kernel<HDIM, HDIM, false, false, false><<<gblocks, THREADS, lds_128_128, stream>>>(
      node_rep, W2, P2, N_NODES, nullptr, nullptr, nullptr, nullptr, nullptr);

  // edge messages: stats pass, BN finalize, then recompute + scatter (avoids 328MB intermediate)
  edge_kernel<false><<<eblocks, THREADS, lds_edge, stream>>>(edge_rep, edge_attr, src, dst,
      P1, P2, W3, We, be, nullptr, nullptr, sum1, sq1, Y);
  bn_finalize_kernel<<<1, HDIM, 0, stream>>>(sum1, sq1, bn_g, bn_b,
      1.0f / (float)N_EDGES, HDIM, scale1, shift1);
  edge_kernel<true><<<eblocks, THREADS, lds_edge, stream>>>(edge_rep, edge_attr, src, dst,
      P1, P2, W3, We, be, scale1, shift1, nullptr, nullptr, Y);

  // MLP with BN fused into the consumer's A-fragment load
  gemm_kernel<HDIM, H2DIM, false, true, false><<<gblocks, THREADS, lds_128_256, stream>>>(
      Y, Wm1, H1, N_NODES, nullptr, nullptr, nullptr, sum2, sq2);
  bn_finalize_kernel<<<1, H2DIM, 0, stream>>>(sum2, sq2, g1, b1,
      1.0f / (float)N_NODES, H2DIM, scale2, shift2);
  gemm_kernel<H2DIM, H2DIM, true, true, false><<<gblocks, THREADS, lds_256_256, stream>>>(
      H1, Wm2, H2, N_NODES, scale2, shift2, nullptr, sum3, sq3);
  bn_finalize_kernel<<<1, H2DIM, 0, stream>>>(sum3, sq3, g2, b2,
      1.0f / (float)N_NODES, H2DIM, scale3, shift3);
  gemm_kernel<H2DIM, HDIM, true, false, true><<<gblocks, THREADS, lds_256_128, stream>>>(
      H2, Wm3, (float*)d_out, N_NODES, scale3, shift3, bm3, nullptr, nullptr);
}
